// LinearAttentionBlock_71116068487390
// MI455X (gfx1250) — matrix operand order repaired and checked by reading, not compiled, not run on hardware
//
#include <hip/hip_runtime.h>
#include <hip/hip_bf16.h>

typedef unsigned short u16;
typedef __attribute__((ext_vector_type(16))) __bf16 v16bf;
typedef __attribute__((ext_vector_type(8)))  float  v8f;

#define BDIM 4
#define NSEQ 4096
#define CDIM 1024
#define HEADS 16
#define HD 64
#define HID 4096
#define MROWS (BDIM * NSEQ)   // 16384

// CDNA5 async LDS-DMA path (ASYNCcnt) if toolchain exposes the builtins.
#if defined(__AMDGCN__) && \
    __has_builtin(__builtin_amdgcn_global_load_async_to_lds_b128) && \
    __has_builtin(__builtin_amdgcn_s_wait_asynccnt)
#define USE_ASYNC 1
#else
#define USE_ASYNC 0
#endif

union FragU { uint4 u[2]; v16bf v; };

__device__ inline u16 f2bf(float f) {
  unsigned u = __float_as_uint(f);
  u += 0x7fffu + ((u >> 16) & 1u);
  return (u16)(u >> 16);
}
__device__ inline float bf2f(u16 h) { return __uint_as_float(((unsigned)h) << 16); }

__device__ inline v8f wmma_bf16(v16bf a, v16bf b, v8f c) {
  // D = A(16x32 bf16) * B(32x16 bf16) + C(16x16 f32)
  return __builtin_amdgcn_wmma_f32_16x16x32_bf16(false, a, false, b, (short)0, c,
                                                 false, false);
}

// A-fragment (16x32, MxK): lane l -> row m = l&15;
// element e -> K = (e/8)*16 + (l>>4)*8 + (e&7)  => two contiguous 8-elem runs.
__device__ inline v16bf frag_a(const u16* lds, int row, int stride, int lane) {
  const u16* p = lds + row * stride + ((lane >> 4) << 3);
  FragU f;
  f.u[0] = *(const uint4*)(p);
  f.u[1] = *(const uint4*)(p + 16);
  return f.v;
}
// B-fragment (32x16, KxN): lane l -> col n = l&15;
// element e -> K = (e/8)*16 + (l>>4)*8 + (e&7)  => two contiguous 8-elem runs.
__device__ inline v16bf frag_b(const u16* lds, int col, int stride, int lane) {
  const u16* p = lds + col * stride + ((lane >> 4) << 3);
  FragU f;
  f.u[0] = *(const uint4*)(p);
  f.u[1] = *(const uint4*)(p + 16);
  return f.v;
}

#if USE_ASYNC
// Builtin signature (from clang diagnostic): params are pointers to a
// vector_size(16) int in AS(1)/AS(3), plus imm offset and imm cpol.
typedef int v4i __attribute__((vector_size(16)));
typedef __attribute__((address_space(1))) v4i* gv4i_p;
typedef __attribute__((address_space(3))) v4i* lv4i_p;

// Generic LDS pointer -> AS(3): low 32 bits of a generic LDS address are the
// LDS offset on AMDGPU. Generic global pointer -> AS(1): identity.
__device__ inline void async_ld128(const u16* g, u16* l) {
  __builtin_amdgcn_global_load_async_to_lds_b128(
      (gv4i_p)(uintptr_t)g, (lv4i_p)(unsigned)(uintptr_t)l, 0, 0);
}
#endif

// ---------------------------------------------------------------- converts
__global__ __launch_bounds__(256) void cvt_f32_bf16(const float* __restrict__ in,
                                                    u16* __restrict__ out, int n) {
  int i = blockIdx.x * 256 + threadIdx.x;
  if (i < n) out[i] = f2bf(in[i]);
}

// ---------------------------------------------------------------- layernorm
__global__ __launch_bounds__(256) void ln_kernel(const float* __restrict__ x,
                                                 const float* __restrict__ w,
                                                 const float* __restrict__ b,
                                                 u16* __restrict__ out) {
  __shared__ float r1[256], r2[256];
  int row = blockIdx.x, t = threadIdx.x;
  const float* xr = x + (size_t)row * CDIM;
  float v[4], s = 0.f, s2 = 0.f;
#pragma unroll
  for (int i = 0; i < 4; ++i) {
    float f = xr[t + 256 * i];
    v[i] = f; s += f; s2 += f * f;
  }
  r1[t] = s; r2[t] = s2;
  __syncthreads();
  for (int off = 128; off > 0; off >>= 1) {
    if (t < off) { r1[t] += r1[t + off]; r2[t] += r2[t + off]; }
    __syncthreads();
  }
  float mu  = r1[0] * (1.f / CDIM);
  float var = r2[0] * (1.f / CDIM) - mu * mu;
  float inv = rsqrtf(var + 1e-5f);
  u16* orow = out + (size_t)row * CDIM;
#pragma unroll
  for (int i = 0; i < 4; ++i) {
    int c = t + 256 * i;
    orow[c] = f2bf((v[i] - mu) * inv * w[c] + b[c]);
  }
}

// ---------------------------------------------------------------- big GEMM
// out[M,Nc] = A[M,K](bf16) @ W[Nc,K]^T(bf16) + bias, fused epilogue.
// 128x128 block tile, K-step 32, 8 waves; wave tile 32x64 = 2x4 WMMA accums.
// Async path: double-buffered LDS fed by global_load_async_to_lds_b128,
// one barrier per K-step, DMA overlapped with WMMA.
#define EPI_SILU_QKV 0
#define EPI_GELU     1
#define EPI_RES_F32  2

template <int EPI>
__global__ __launch_bounds__(256) void gemm_kernel(
    const u16* __restrict__ A, const u16* __restrict__ W,
    const float* __restrict__ bias, const float* __restrict__ resid,
    u16* __restrict__ outb, float* __restrict__ outf, int M, int Nc, int K) {
  __shared__ u16 As[2][128 * 40];
  __shared__ u16 Bs[2][128 * 40];
  int tid = threadIdx.x, lane = tid & 31, wid = tid >> 5;
  int bm = blockIdx.y * 128, bn = blockIdx.x * 128;
  int wm = (wid & 3) * 32, wn = (wid >> 2) * 64;

  v8f acc[2][4];
#pragma unroll
  for (int i = 0; i < 2; ++i)
#pragma unroll
    for (int j = 0; j < 4; ++j)
#pragma unroll
      for (int r = 0; r < 8; ++r) acc[i][j][r] = 0.f;

  int lrow = tid >> 1, lk = (tid & 1) * 16;
  const u16* Ag = A + (size_t)(bm + lrow) * K + lk;
  const u16* Wg = W + (size_t)(bn + lrow) * K + lk;
  int loff = lrow * 40 + lk;
  const int nsteps = K >> 5;

#if USE_ASYNC
  // ---- async double-buffered pipeline ----
  async_ld128(Ag,     &As[0][loff]);
  async_ld128(Ag + 8, &As[0][loff + 8]);
  async_ld128(Wg,     &Bs[0][loff]);
  async_ld128(Wg + 8, &Bs[0][loff + 8]);
  __builtin_amdgcn_s_wait_asynccnt(0);
  __syncthreads();
  for (int step = 0; step < nsteps; ++step) {
    int cur = step & 1, nxt = cur ^ 1;
    if (step + 1 < nsteps) {
      // safe: end-of-previous-step barrier guarantees buf[nxt] readers done
      int k0 = (step + 1) << 5;
      async_ld128(Ag + k0,     &As[nxt][loff]);
      async_ld128(Ag + k0 + 8, &As[nxt][loff + 8]);
      async_ld128(Wg + k0,     &Bs[nxt][loff]);
      async_ld128(Wg + k0 + 8, &Bs[nxt][loff + 8]);
    }
    v16bf af[2], bf_[4];
#pragma unroll
    for (int mi = 0; mi < 2; ++mi)
      af[mi] = frag_a(As[cur], wm + mi * 16 + (lane & 15), 40, lane);
#pragma unroll
    for (int ni = 0; ni < 4; ++ni)
      bf_[ni] = frag_b(Bs[cur], wn + ni * 16 + (lane & 15), 40, lane);
#pragma unroll
    for (int mi = 0; mi < 2; ++mi)
#pragma unroll
      for (int ni = 0; ni < 4; ++ni)
        acc[mi][ni] = wmma_bf16(af[mi], bf_[ni], acc[mi][ni]);
    __builtin_amdgcn_s_wait_asynccnt(0);
    __syncthreads();
  }
#else
  // ---- fallback: reg staging, next-tile loads overlapped with WMMA ----
  uint4 a0 = *(const uint4*)(Ag);
  uint4 a1 = *(const uint4*)(Ag + 8);
  uint4 b0 = *(const uint4*)(Wg);
  uint4 b1 = *(const uint4*)(Wg + 8);
  for (int step = 0; step < nsteps; ++step) {
    __syncthreads();                    // prior-step LDS readers done
    *(uint4*)(&As[0][loff])     = a0;
    *(uint4*)(&As[0][loff + 8]) = a1;
    *(uint4*)(&Bs[0][loff])     = b0;
    *(uint4*)(&Bs[0][loff + 8]) = b1;
    __syncthreads();
    if (step + 1 < nsteps) {            // in flight during the WMMAs below
      int k0 = (step + 1) << 5;
      __builtin_prefetch(Ag + k0 + 32, 0, 1);
      __builtin_prefetch(Wg + k0 + 32, 0, 1);
      a0 = *(const uint4*)(Ag + k0);
      a1 = *(const uint4*)(Ag + k0 + 8);
      b0 = *(const uint4*)(Wg + k0);
      b1 = *(const uint4*)(Wg + k0 + 8);
    }
    v16bf af[2], bf_[4];
#pragma unroll
    for (int mi = 0; mi < 2; ++mi)
      af[mi] = frag_a(As[0], wm + mi * 16 + (lane & 15), 40, lane);
#pragma unroll
    for (int ni = 0; ni < 4; ++ni)
      bf_[ni] = frag_b(Bs[0], wn + ni * 16 + (lane & 15), 40, lane);
#pragma unroll
    for (int mi = 0; mi < 2; ++mi)
#pragma unroll
      for (int ni = 0; ni < 4; ++ni)
        acc[mi][ni] = wmma_bf16(af[mi], bf_[ni], acc[mi][ni]);
  }
#endif

#pragma unroll
  for (int mi = 0; mi < 2; ++mi)
#pragma unroll
    for (int ni = 0; ni < 4; ++ni)
#pragma unroll
      for (int r = 0; r < 8; ++r) {
        int rg = bm + wm + mi * 16 + ((lane >> 4) << 3) + r;
        int cg = bn + wn + ni * 16 + (lane & 15);
        float v = acc[mi][ni][r] + bias[cg];
        if (EPI == EPI_SILU_QKV) {
          if ((cg % 192) < 128) v = v / (1.f + expf(-v));   // silu on q,k
          outb[(size_t)rg * Nc + cg] = f2bf(v);
        } else if (EPI == EPI_GELU) {
          v = 0.5f * v * (1.f + erff(v * 0.70710678f));
          outb[(size_t)rg * Nc + cg] = f2bf(v);
        } else {                                            // bias+residual f32
          v += resid[(size_t)rg * Nc + cg];
          outf[(size_t)rg * Nc + cg] = v;
        }
      }
}

// ---------------------------------------------------------------- state = k^T v
__global__ __launch_bounds__(256) void state_kernel(const u16* __restrict__ qkv,
                                                    u16* __restrict__ state_t,
                                                    float* __restrict__ ksum) {
  __shared__ u16 kt[64 * 40];   // kt[d][n]
  __shared__ u16 vt[64 * 40];   // vt[e][n]
  __shared__ float red[256];
  int bh = blockIdx.x, b = bh >> 4, h = bh & 15;
  int tid = threadIdx.x, lane = tid & 31, wid = tid >> 5;
  int mt = (wid & 3) * 16;        // d-tile
  int nt = (wid >> 2) * 32;       // e-tile base (2 tiles per wave)
  size_t rowbase = (size_t)b * NSEQ * (3 * CDIM);
  int kcol = h * 192 + 64, vcol = h * 192 + 128;

  v8f acc[2];
#pragma unroll
  for (int j = 0; j < 2; ++j)
#pragma unroll
    for (int r = 0; r < 8; ++r) acc[j][r] = 0.f;

  int sn = tid >> 3, d0 = (tid & 7) * 8;   // stage: 32 n-rows x 8 d each
  for (int n0 = 0; n0 < NSEQ; n0 += 32) {
    size_t g = rowbase + (size_t)(n0 + sn) * (3 * CDIM);
    uint4 kv = *(const uint4*)(qkv + g + kcol + d0);
    uint4 vv = *(const uint4*)(qkv + g + vcol + d0);
    __syncthreads();
    const u16* kp = (const u16*)&kv;
    const u16* vp = (const u16*)&vv;
#pragma unroll
    for (int i = 0; i < 8; ++i) {
      kt[(d0 + i) * 40 + sn] = kp[i];
      vt[(d0 + i) * 40 + sn] = vp[i];
    }
    __syncthreads();
    v16bf fa = frag_a(kt, mt + (lane & 15), 40, lane);
    v16bf f0 = frag_b(vt, nt + (lane & 15), 40, lane);
    v16bf f1 = frag_b(vt, nt + 16 + (lane & 15), 40, lane);
    acc[0] = wmma_bf16(fa, f0, acc[0]);
    acc[1] = wmma_bf16(fa, f1, acc[1]);
  }

  // store transposed: state_t[bh][e][d]
#pragma unroll
  for (int j = 0; j < 2; ++j)
#pragma unroll
    for (int r = 0; r < 8; ++r) {
      int d = mt + ((lane >> 4) << 3) + r;
      int e = nt + j * 16 + (lane & 15);
      state_t[(size_t)bh * (HD * HD) + e * HD + d] = f2bf(acc[j][r]);
    }

  // k_sum[d] = sum_n k[n][d]
  int dd = tid & 63, qq = tid >> 6;
  float s = 0.f;
  for (int n = qq; n < NSEQ; n += 4)
    s += bf2f(qkv[rowbase + (size_t)n * (3 * CDIM) + kcol + dd]);
  red[tid] = s;
  __syncthreads();
  if (qq == 0)
    ksum[bh * HD + dd] = red[dd] + red[64 + dd] + red[128 + dd] + red[192 + dd];
}

// ---------------------------------------------------------------- out = (q @ state) / (q.k_sum + eps)
__global__ __launch_bounds__(256) void attn_kernel(const u16* __restrict__ qkv,
                                                   const u16* __restrict__ state_t,
                                                   const float* __restrict__ ksum,
                                                   u16* __restrict__ attn) {
  __shared__ float den[128];
  __shared__ float ks[64];
  int blk = blockIdx.x;
  int bh = blk >> 5, ntile = blk & 31;
  int b = bh >> 4, h = bh & 15;
  int nbase = ntile * 128;
  int tid = threadIdx.x, lane = tid & 31, wid = tid >> 5;
  size_t rowbase = (size_t)b * NSEQ * (3 * CDIM);
  int qcol = h * 192;

  if (tid < 64) ks[tid] = ksum[bh * HD + tid];
  __syncthreads();
  if (tid < 128) {
    const u16* qr = qkv + rowbase + (size_t)(nbase + tid) * (3 * CDIM) + qcol;
    float s = 0.f;
#pragma unroll 8
    for (int d = 0; d < HD; ++d) s += bf2f(qr[d]) * ks[d];
    den[tid] = s;
  }
  __syncthreads();

  v8f acc[4];
#pragma unroll
  for (int j = 0; j < 4; ++j)
#pragma unroll
    for (int r = 0; r < 8; ++r) acc[j][r] = 0.f;

  const u16* st = state_t + (size_t)bh * (HD * HD);
  const u16* qg =
      qkv + rowbase + (size_t)(nbase + wid * 16 + (lane & 15)) * (3 * CDIM) + qcol;
#pragma unroll
  for (int s = 0; s < 2; ++s) {          // K = 64 -> two 32-steps
    FragU fa;
    fa.u[0] = *(const uint4*)(qg + s * 32 + ((lane >> 4) << 3));
    fa.u[1] = *(const uint4*)(qg + s * 32 + 16 + ((lane >> 4) << 3));
#pragma unroll
    for (int nt = 0; nt < 4; ++nt) {
      int e = nt * 16 + (lane & 15);
      const u16* sp = st + e * HD + s * 32 + ((lane >> 4) << 3);
      FragU fb;
      fb.u[0] = *(const uint4*)(sp);
      fb.u[1] = *(const uint4*)(sp + 16);
      acc[nt] = wmma_bf16(fa.v, fb.v, acc[nt]);
    }
  }

#pragma unroll
  for (int nt = 0; nt < 4; ++nt)
#pragma unroll
    for (int r = 0; r < 8; ++r) {
      int nrow = wid * 16 + ((lane >> 4) << 3) + r;
      int e = nt * 16 + (lane & 15);
      float dv = den[nrow] + 1e-6f;
      attn[(size_t)(b * NSEQ + nbase + nrow) * CDIM + h * HD + e] =
          f2bf(acc[nt][r] / dv);
    }
}

// ---------------------------------------------------------------- launch
extern "C" void kernel_launch(void* const* d_in, const int* in_sizes, int n_in,
                              void* d_out, int out_size, void* d_ws, size_t ws_size,
                              hipStream_t stream) {
  const float* x     = (const float*)d_in[0];
  const float* ln1_w = (const float*)d_in[1];
  const float* ln1_b = (const float*)d_in[2];
  const float* ln2_w = (const float*)d_in[3];
  const float* ln2_b = (const float*)d_in[4];
  const float* qkv_w = (const float*)d_in[5];
  const float* qkv_b = (const float*)d_in[6];
  const float* out_w = (const float*)d_in[7];
  const float* out_b = (const float*)d_in[8];
  const float* fc1_w = (const float*)d_in[9];
  const float* fc1_b = (const float*)d_in[10];
  const float* fc2_w = (const float*)d_in[11];
  const float* fc2_b = (const float*)d_in[12];

  char* p = (char*)d_ws;
  auto take = [&](size_t bytes) {
    char* r = p;
    p += (bytes + 255) & ~size_t(255);
    return r;
  };
  u16*   xln     = (u16*)  take((size_t)MROWS * CDIM * 2);
  u16*   qkv     = (u16*)  take((size_t)MROWS * 3 * CDIM * 2);
  u16*   wqkv    = (u16*)  take((size_t)3 * CDIM * CDIM * 2);
  u16*   wout    = (u16*)  take((size_t)CDIM * CDIM * 2);
  u16*   wfc1    = (u16*)  take((size_t)HID * CDIM * 2);
  u16*   wfc2    = (u16*)  take((size_t)CDIM * HID * 2);
  u16*   state_t = (u16*)  take((size_t)BDIM * HEADS * HD * HD * 2);
  float* ksum    = (float*)take((size_t)BDIM * HEADS * HD * 4);
  u16*   attn    = (u16*)  take((size_t)MROWS * CDIM * 2);
  float* x1      = (float*)take((size_t)MROWS * CDIM * 4);
  u16*   hln     = (u16*)  take((size_t)MROWS * CDIM * 2);
  u16*   h1      = (u16*)  take((size_t)MROWS * HID * 2);

  cvt_f32_bf16<<<(3 * CDIM * CDIM + 255) / 256, 256, 0, stream>>>(qkv_w, wqkv, 3 * CDIM * CDIM);
  cvt_f32_bf16<<<(CDIM * CDIM + 255) / 256, 256, 0, stream>>>(out_w, wout, CDIM * CDIM);
  cvt_f32_bf16<<<(HID * CDIM + 255) / 256, 256, 0, stream>>>(fc1_w, wfc1, HID * CDIM);
  cvt_f32_bf16<<<(CDIM * HID + 255) / 256, 256, 0, stream>>>(fc2_w, wfc2, CDIM * HID);

  ln_kernel<<<MROWS, 256, 0, stream>>>(x, ln1_w, ln1_b, xln);
  gemm_kernel<EPI_SILU_QKV><<<dim3(3 * CDIM / 128, MROWS / 128), 256, 0, stream>>>(
      xln, wqkv, qkv_b, nullptr, qkv, nullptr, MROWS, 3 * CDIM, CDIM);
  state_kernel<<<BDIM * HEADS, 256, 0, stream>>>(qkv, state_t, ksum);
  attn_kernel<<<BDIM * HEADS * (NSEQ / 128), 256, 0, stream>>>(qkv, state_t, ksum, attn);
  gemm_kernel<EPI_RES_F32><<<dim3(CDIM / 128, MROWS / 128), 256, 0, stream>>>(
      attn, wout, out_b, x, nullptr, x1, MROWS, CDIM, CDIM);
  ln_kernel<<<MROWS, 256, 0, stream>>>(x1, ln2_w, ln2_b, hln);
  gemm_kernel<EPI_GELU><<<dim3(HID / 128, MROWS / 128), 256, 0, stream>>>(
      hln, wfc1, fc1_b, nullptr, h1, nullptr, MROWS, HID, CDIM);
  gemm_kernel<EPI_RES_F32><<<dim3(CDIM / 128, MROWS / 128), 256, 0, stream>>>(
      h1, wfc2, fc2_b, x1, nullptr, (float*)d_out, MROWS, CDIM, HID);
}
